// decoder_10144712753517
// MI455X (gfx1250) — compile-verified, matrix operand order checked by reading, not run on hardware
//
#include <hip/hip_runtime.h>
#include <math.h>

// ---------------- problem constants (from reference) ----------------
constexpr int NN    = 20000;
constexpr int EE    = 100000;
constexpr int IN_C  = 128;
constexpr int DD    = 32;
constexpr int HH    = 32;
constexpr int HDHD  = 1024;   // H * D
constexpr int UU    = 2;
constexpr float EPSF        = 1e-5f;
constexpr float SM_EPSF     = 1e-16f;
constexpr float INV_SQRT_DF = 0.17677669529663687f; // 1/sqrt(32)

// ---------------- WMMA types ----------------
typedef __attribute__((ext_vector_type(16))) __bf16 v16bf;
typedef __attribute__((ext_vector_type(8)))  float  v8f;

__device__ __forceinline__ void atomicMaxF(float* addr, float val) {
  // IEEE ordering trick; buffer is initialized to -inf
  if (val >= 0.0f) atomicMax((int*)addr, __float_as_int(val));
  else             atomicMin((unsigned int*)addr, __float_as_uint(val));
}

// ---------------- fragment packing (fp32 -> bf16, WMMA lane layout) ----------
// A[M x K] -> Ap[tm][ks][lane][16] ; element i of lane l: row=tm*16+(l&15),
// kk = ks*32 + ((l>>4)*8) + (i<8 ? i : i+8)   (ISA 7.12.2 16-bit A layout)
__global__ void pack_a_kernel(const float* __restrict__ A, unsigned short* __restrict__ Ap,
                              int M, int K) {
  int i = blockIdx.x * blockDim.x + threadIdx.x;
  if (i >= M * K) return;
  const int elem = i & 15;
  const int lane = (i >> 4) & 31;
  const int rest = i >> 9;              // tm * ksteps + ks
  const int ksteps = K >> 5;
  const int tm = rest / ksteps;
  const int ks = rest - tm * ksteps;
  const int row = (tm << 4) + (lane & 15);
  const int kk  = (ks << 5) + ((lane >> 4) << 3) + (elem < 8 ? elem : elem + 8);
  union { __bf16 b; unsigned short s; } cv;
  cv.b = (__bf16)A[(size_t)row * K + kk];
  Ap[i] = cv.s;
}

// W[K x Nc] -> Wp[ks][tn][lane][16] ; element i of lane l: col=tn*16+(l&15),
// kk mapping same as A (B mirrors A with lanes holding N).
__global__ void pack_w_kernel(const float* __restrict__ W, unsigned short* __restrict__ Wp,
                              int K, int Nc) {
  int i = blockIdx.x * blockDim.x + threadIdx.x;
  if (i >= K * Nc) return;
  const int elem = i & 15;
  const int lane = (i >> 4) & 31;
  const int rest = i >> 9;              // ks * tilesn + tn
  const int tilesn = Nc >> 4;
  const int ks = rest / tilesn;
  const int tn = rest - ks * tilesn;
  const int col = (tn << 4) + (lane & 15);
  const int kk  = (ks << 5) + ((lane >> 4) << 3) + (elem < 8 ? elem : elem + 8);
  union { __bf16 b; unsigned short s; } cv;
  cv.b = (__bf16)W[(size_t)kk * Nc + col];
  Wp[i] = cv.s;
}

// ---- bf16 WMMA GEMM on pre-packed fragments: C = A @ W (+bias) (+=C) ----
// One wave owns a 16 x (16*TN) output strip; hot loop is pure fragment
// loads + v_wmma (no conversions, no strided loads).
template<int TN, bool ACC, bool BIAS>
__global__ __launch_bounds__(256)
void gemm_bf16_kernel(const unsigned short* __restrict__ Ap,
                      const unsigned short* __restrict__ Wp,
                      const float* __restrict__ bias,
                      float* __restrict__ C,
                      int M, int K, int Nc)
{
  const int lane   = threadIdx.x & 31;
  const int wave   = blockIdx.x * (blockDim.x >> 5) + (threadIdx.x >> 5);
  const int tilesn = Nc >> 4;
  const int strips = tilesn / TN;
  const int total  = (M >> 4) * strips;
  if (wave >= total) return;                 // wave-uniform: EXEC stays all-ones
  const int tm = wave / strips;
  const int ts = wave - tm * strips;

  const int ksteps = K >> 5;
  const unsigned short* ApL = Ap + (((size_t)tm * ksteps) * 32 + lane) * 16;
  const unsigned short* WpL = Wp + ((size_t)lane) * 16;

  v8f c[TN];
#pragma unroll
  for (int t = 0; t < TN; ++t) c[t] = (v8f){0.f,0.f,0.f,0.f,0.f,0.f,0.f,0.f};

  for (int ks = 0; ks < ksteps; ++ks) {
    const v16bf av = *(const v16bf*)(ApL + (size_t)ks * 512);   // 32B fragment
#pragma unroll
    for (int t = 0; t < TN; ++t) {
      const int tile = ts * TN + t;
      const v16bf bv = *(const v16bf*)(WpL + ((size_t)ks * tilesn + tile) * 512);
      c[t] = __builtin_amdgcn_wmma_f32_16x16x32_bf16(false, av, false, bv,
                                                     (short)0, c[t], false, false);
    }
  }

  const int l15 = lane & 15;
  const int m0  = (tm << 4) + ((lane >> 4) << 3);
#pragma unroll
  for (int t = 0; t < TN; ++t) {
    const int ncol = ((ts * TN + t) << 4) + l15;
    float ba = 0.0f;
    if (BIAS) ba = bias[ncol];
#pragma unroll
    for (int r = 0; r < 8; ++r) {
      float val = c[t][r] + ba;
      float* p = C + (size_t)(m0 + r) * Nc + ncol;
      if (ACC) val += *p;
      *p = val;
    }
  }
}

// ---------------- elementwise / graph kernels ----------------
__global__ void fill_kernel(float* __restrict__ p, float v, int n) {
  int i = blockIdx.x * blockDim.x + threadIdx.x;
  if (i < n) p[i] = v;
}

__global__ void count_kernel(const int* __restrict__ dst, float* __restrict__ cnt, int E) {
  int i = blockIdx.x * blockDim.x + threadIdx.x;
  if (i < E) atomicAdd(&cnt[dst[i]], 1.0f);
}

__global__ void scatter_feat_kernel(const float* __restrict__ x,
                                    const int* __restrict__ src,
                                    const int* __restrict__ dst,
                                    float* __restrict__ s, int E, int C) {
  int i = blockIdx.x * blockDim.x + threadIdx.x;
  if (i >= E * C) return;
  const int e = i / C, c = i - e * C;
  atomicAdd(&s[(size_t)dst[e] * C + c], x[(size_t)src[e] * C + c]);
}

__global__ void mean_div_kernel(float* __restrict__ s, const float* __restrict__ cnt,
                                int n, int C) {
  int i = blockIdx.x * blockDim.x + threadIdx.x;
  if (i >= n * C) return;
  s[i] /= fmaxf(cnt[i / C], 1.0f);
}

__global__ void edge_score_kernel(const float* __restrict__ q, const float* __restrict__ k,
                                  const int* __restrict__ src, const int* __restrict__ dst,
                                  float* __restrict__ sc, float* __restrict__ m, int E) {
  int i = blockIdx.x * blockDim.x + threadIdx.x;
  if (i >= E * HH) return;
  const int e = i >> 5, h = i & 31;
  const int sn = src[e], dn = dst[e];
  const float4* qp = (const float4*)(q + (size_t)dn * HDHD + h * DD);
  const float4* kp = (const float4*)(k + (size_t)sn * HDHD + h * DD);
  float a = 0.f;
#pragma unroll
  for (int j = 0; j < 8; ++j) {
    const float4 x4 = qp[j], y4 = kp[j];
    a += x4.x * y4.x + x4.y * y4.y + x4.z * y4.z + x4.w * y4.w;
  }
  a *= INV_SQRT_DF;
  sc[i] = a;
  atomicMaxF(m + (size_t)dn * HH + h, a);
}

__global__ void edge_exp_kernel(float* __restrict__ sc, const float* __restrict__ m,
                                float* __restrict__ ss, const int* __restrict__ dst, int E) {
  int i = blockIdx.x * blockDim.x + threadIdx.x;
  if (i >= E * HH) return;
  const int e = i >> 5, h = i & 31;
  const int dn = dst[e];
  const float v = expf(sc[i] - m[(size_t)dn * HH + h]);
  sc[i] = v;
  atomicAdd(ss + (size_t)dn * HH + h, v);
}

__global__ void edge_msg_kernel(const float* __restrict__ sc, const float* __restrict__ ss,
                                const float* __restrict__ v, const int* __restrict__ src,
                                const int* __restrict__ dst, float* __restrict__ O, int E) {
  int i = blockIdx.x * blockDim.x + threadIdx.x;
  if (i >= E * HH) return;
  const int e = i >> 5, h = i & 31;
  const int sn = src[e], dn = dst[e];
  const float alpha = sc[i] / (ss[(size_t)dn * HH + h] + SM_EPSF);
  const float* vp = v + (size_t)sn * HDHD + h * DD;
  float* op = O + (size_t)dn * HDHD + h * DD;
#pragma unroll
  for (int d = 0; d < DD; d += 4) {
    const float4 t = *(const float4*)(vp + d);
    atomicAdd(op + d + 0, alpha * t.x);
    atomicAdd(op + d + 1, alpha * t.y);
    atomicAdd(op + d + 2, alpha * t.z);
    atomicAdd(op + d + 3, alpha * t.w);
  }
}

__global__ void gln_reduce_kernel(const float* __restrict__ x, long long n,
                                  float* __restrict__ acc) {
  long long i = (long long)blockIdx.x * blockDim.x + threadIdx.x;
  const long long stride = (long long)gridDim.x * blockDim.x;
  float s = 0.f, s2 = 0.f;
  for (; i < n; i += stride) { const float t = x[i]; s += t; s2 += t * t; }
#pragma unroll
  for (int off = 16; off > 0; off >>= 1) {
    s  += __shfl_xor(s,  off, 32);
    s2 += __shfl_xor(s2, off, 32);
  }
  if ((threadIdx.x & 31) == 0) { atomicAdd(&acc[0], s); atomicAdd(&acc[1], s2); }
}

__global__ void gln_apply_kernel(const float* __restrict__ x, const float* __restrict__ acc,
                                 const float* __restrict__ w, const float* __restrict__ b,
                                 float* __restrict__ y, int n, int C) {
  int i = blockIdx.x * blockDim.x + threadIdx.x;
  if (i >= n) return;
  const float inv_n = 1.0f / (float)n;
  const float mean  = acc[0] * inv_n;
  const float var   = fmaxf(acc[1] * inv_n - mean * mean, 0.0f);
  const float inv   = 1.0f / (sqrtf(var) + EPSF);
  const int c = i % C;
  y[i] = (x[i] - mean) * inv * w[c] + b[c];
}

// ---------------- host orchestration ----------------
static inline void launch_gemm(const float* A, const float* W, const float* bias,
                               float* C, int M, int K, int Nc, bool accum,
                               unsigned short* apack, unsigned short* wpack,
                               hipStream_t st) {
  pack_a_kernel<<<(M * K + 255) / 256, 256, 0, st>>>(A, apack, M, K);
  pack_w_kernel<<<(K * Nc + 255) / 256, 256, 0, st>>>(W, wpack, K, Nc);
  if ((Nc & 63) == 0) {                               // Nc = 1024: 16x64 strips
    const int total  = (M >> 4) * ((Nc >> 4) / 4);
    const int blocks = (total + 7) / 8;
    if (bias) {
      if (accum) gemm_bf16_kernel<4, true,  true ><<<blocks, 256, 0, st>>>(apack, wpack, bias, C, M, K, Nc);
      else       gemm_bf16_kernel<4, false, true ><<<blocks, 256, 0, st>>>(apack, wpack, bias, C, M, K, Nc);
    } else {
      if (accum) gemm_bf16_kernel<4, true,  false><<<blocks, 256, 0, st>>>(apack, wpack, bias, C, M, K, Nc);
      else       gemm_bf16_kernel<4, false, false><<<blocks, 256, 0, st>>>(apack, wpack, bias, C, M, K, Nc);
    }
  } else {                                            // Nc = 32: 16x32 strips
    const int total  = (M >> 4) * ((Nc >> 4) / 2);
    const int blocks = (total + 7) / 8;
    if (bias) {
      if (accum) gemm_bf16_kernel<2, true,  true ><<<blocks, 256, 0, st>>>(apack, wpack, bias, C, M, K, Nc);
      else       gemm_bf16_kernel<2, false, true ><<<blocks, 256, 0, st>>>(apack, wpack, bias, C, M, K, Nc);
    } else {
      if (accum) gemm_bf16_kernel<2, true,  false><<<blocks, 256, 0, st>>>(apack, wpack, bias, C, M, K, Nc);
      else       gemm_bf16_kernel<2, false, false><<<blocks, 256, 0, st>>>(apack, wpack, bias, C, M, K, Nc);
    }
  }
}

extern "C" void kernel_launch(void* const* d_in, const int* in_sizes, int n_in,
                              void* d_out, int out_size, void* d_ws, size_t ws_size,
                              hipStream_t stream)
{
  (void)in_sizes; (void)n_in; (void)out_size; (void)ws_size;

  // inputs in setup_inputs() dict order
  const float* x     = (const float*)d_in[0];
  const float* llc   = (const float*)d_in[1];
  const int*   ei    = (const int*)  d_in[2];
  const float* ew_l  = (const float*)d_in[3];
  const float* eb_l  = (const float*)d_in[4];
  const float* ew_r  = (const float*)d_in[5];
  const float* a1_wq = (const float*)d_in[6];  const float* a2_wq = (const float*)d_in[7];
  const float* a1_bq = (const float*)d_in[8];  const float* a2_bq = (const float*)d_in[9];
  const float* a1_wk = (const float*)d_in[10]; const float* a2_wk = (const float*)d_in[11];
  const float* a1_bk = (const float*)d_in[12]; const float* a2_bk = (const float*)d_in[13];
  const float* a1_wv = (const float*)d_in[14]; const float* a2_wv = (const float*)d_in[15];
  const float* a1_bv = (const float*)d_in[16]; const float* a2_bv = (const float*)d_in[17];
  const float* a1_ws = (const float*)d_in[18]; const float* a2_ws = (const float*)d_in[19];
  const float* a1_bs = (const float*)d_in[20]; const float* a2_bs = (const float*)d_in[21];
  const float* c1_w  = (const float*)d_in[22]; const float* c1_b  = (const float*)d_in[23];
  const float* c2_w  = (const float*)d_in[24]; const float* c2_b  = (const float*)d_in[25];
  const float* n1_w  = (const float*)d_in[26]; const float* n1_b  = (const float*)d_in[27];
  const float* n2_w  = (const float*)d_in[28]; const float* n2_b  = (const float*)d_in[29];
  const float* n3_w  = (const float*)d_in[30]; const float* n3_b  = (const float*)d_in[31];
  const float* s_wl  = (const float*)d_in[32]; const float* s_bl  = (const float*)d_in[33];
  const float* s_wr  = (const float*)d_in[34];

  const int* src = ei;
  const int* dst = ei + EE;

  // workspace layout (floats); total ~367 MB
  float* ws = (float*)d_ws;
  size_t o = 0;
  float* h    = ws + o; o += (size_t)NN * DD;
  float* t0   = ws + o; o += (size_t)NN * DD;
  float* cnt  = ws + o; o += (size_t)NN;
  float* sag  = ws + o; o += (size_t)NN * IN_C;
  float* q    = ws + o; o += (size_t)NN * HDHD;
  float* k    = ws + o; o += (size_t)NN * HDHD;
  float* v    = ws + o; o += (size_t)NN * HDHD;
  float* O    = ws + o; o += (size_t)NN * HDHD;
  float* sc   = ws + o; o += (size_t)EE * HH;
  float* m    = ws + o; o += (size_t)NN * HH;
  float* ssum = ws + o; o += (size_t)NN * HH;
  float* acc  = ws + o; o += 8;
  unsigned short* apack = (unsigned short*)(ws + o); o += (size_t)NN * IN_C / 2;  // bf16 A
  unsigned short* wpack = (unsigned short*)(ws + o); o += (size_t)IN_C * HDHD / 2; // bf16 W

  auto fill = [&](float* p, float val, int n) {
    fill_kernel<<<(n + 255) / 256, 256, 0, stream>>>(p, val, n);
  };

  // incoming-edge counts (constant across all SAGE calls)
  fill(cnt, 0.f, NN);
  count_kernel<<<(EE + 255) / 256, 256, 0, stream>>>(dst, cnt, EE);

  auto gemm = [&](const float* A, const float* W, const float* bias, float* C,
                  int M, int K, int Nc, bool accum) {
    launch_gemm(A, W, bias, C, M, K, Nc, accum, apack, wpack, stream);
  };

  auto sage = [&](const float* xin, int C, const float* wl, const float* bl,
                  const float* wr, float* outp) {
    fill(sag, 0.f, NN * C);
    scatter_feat_kernel<<<(EE * C + 255) / 256, 256, 0, stream>>>(xin, src, dst, sag, EE, C);
    mean_div_kernel<<<(NN * C + 255) / 256, 256, 0, stream>>>(sag, cnt, NN, C);
    gemm(sag, wl, bl,      outp, NN, C, DD, false);
    gemm(xin, wr, nullptr, outp, NN, C, DD, true);
  };

  auto gln = [&](float* xp, const float* w, const float* b) {
    fill(acc, 0.f, 2);
    gln_reduce_kernel<<<1024, 256, 0, stream>>>(xp, (long long)NN * DD, acc);
    gln_apply_kernel<<<(NN * DD + 255) / 256, 256, 0, stream>>>(xp, acc, w, b, xp, NN * DD, DD);
  };

  auto tconv = [&](const float* xsrc, const float* xdst,
                   const float* wq_, const float* bq_, const float* wk_, const float* bk_,
                   const float* wv_, const float* bv_, const float* wsk, const float* bsk) {
    gemm(xdst, wq_, bq_, q, NN, DD, HDHD, false);
    gemm(xsrc, wk_, bk_, k, NN, DD, HDHD, false);
    gemm(xsrc, wv_, bv_, v, NN, DD, HDHD, false);
    gemm(xdst, wsk, bsk, O, NN, DD, HDHD, false);   // skip; msgs add on top
    fill(m,    -INFINITY, NN * HH);
    fill(ssum, 0.f,       NN * HH);
    const int eh = EE * HH;
    edge_score_kernel<<<(eh + 255) / 256, 256, 0, stream>>>(q, k, src, dst, sc, m, EE);
    edge_exp_kernel  <<<(eh + 255) / 256, 256, 0, stream>>>(sc, m, ssum, dst, EE);
    edge_msg_kernel  <<<(eh + 255) / 256, 256, 0, stream>>>(sc, ssum, v, src, dst, O, EE);
  };

  // dec_embed: h = SAGE(x)
  sage(x, IN_C, ew_l, eb_l, ew_r, h);

  for (int u = 0; u < UU; ++u) {
    const size_t wofs = (size_t)u * DD * HDHD;   // (U, D, HD)
    const size_t bofs = (size_t)u * HDHD;        // (U, HD)
    const size_t cofs = (size_t)u * HDHD * DD;   // (U, HD, D)
    const size_t dofs = (size_t)u * DD;          // (U, D)
    const size_t sofs = (size_t)u * DD * DD;     // (U, D, D)

    // attention block: self TransformerConv -> concat proj -> graph LN
    tconv(h, h,
          a1_wq + wofs, a1_bq + bofs, a1_wk + wofs, a1_bk + bofs,
          a1_wv + wofs, a1_bv + bofs, a1_ws + wofs, a1_bs + bofs);
    gemm(O, c1_w + cofs, c1_b + dofs, t0, NN, HDHD, DD, false);
    gln(t0, n1_w + dofs, n1_b + dofs);

    // cross attention: key/value from t0 (=a), query/skip from llc
    tconv(t0, llc,
          a2_wq + wofs, a2_bq + bofs, a2_wk + wofs, a2_bk + bofs,
          a2_wv + wofs, a2_bv + bofs, a2_ws + wofs, a2_bs + bofs);
    gemm(O, c2_w + cofs, c2_b + dofs, t0, NN, HDHD, DD, false);
    gln(t0, n2_w + dofs, n2_b + dofs);

    // feed_forward: SAGE -> graph LN
    sage(t0, DD, s_wl + sofs, s_bl + dofs, s_wr + sofs, h);
    gln(h, n3_w + dofs, n3_b + dofs);
  }

  hipMemcpyAsync(d_out, h, sizeof(float) * (size_t)NN * DD,
                 hipMemcpyDeviceToDevice, stream);
}
